// CachedMistralSelfAttention_62199716380853
// MI455X (gfx1250) — matrix operand order repaired and checked by reading, not compiled, not run on hardware
//
#include <hip/hip_runtime.h>
#include <hip/hip_bf16.h>

typedef __bf16 bf16_t;
typedef __attribute__((ext_vector_type(16))) __bf16 v16bf;
typedef __attribute__((ext_vector_type(8)))  float  v8f;
typedef __attribute__((ext_vector_type(4)))  unsigned int v4u;
typedef __attribute__((ext_vector_type(8)))  int v8i;
typedef __attribute__((ext_vector_type(4)))  int v4i;

#define T_SEQ    2048
#define HEADS    32
#define HKV_N    8
#define DHEAD    128
#define INNER_N  4096
#define KV_INNER 1024
#define NEGBIG   (-1.0e30f)

#if defined(__gfx1250__) && __has_builtin(__builtin_amdgcn_tensor_load_to_lds) && \
    __has_builtin(__builtin_amdgcn_s_wait_tensorcnt)
#define USE_TDM 1
#else
#define USE_TDM 0
#endif

// ---------------------------------------------------------------------------
// fp32 -> bf16 cast
// ---------------------------------------------------------------------------
__global__ void cast_f32_bf16(const float* __restrict__ src,
                              bf16_t* __restrict__ dst, int n) {
  int i = blockIdx.x * blockDim.x + threadIdx.x;
  if (i < n) dst[i] = (bf16_t)src[i];
}

// ---------------------------------------------------------------------------
// C = A (MxK bf16, row-major) * W^T (W is NxK fp32 row-major)
// Block: 256 threads (8 waves), tile 64(M) x 128(N), BK=32.
// A tile is staged global->LDS with the Tensor Data Mover (one descriptor per
// wave, 8 rows each, tracked by TENSORcnt); W tile is converted fp32->bf16 by
// VALU on the way into LDS, overlapping the DMA.
// Each wave computes a 32x32 sub-tile via 2x2 v_wmma_f32_16x16x32_bf16.
// storeT: write C transposed as bf16 [N][M]. f32out: write fp32 [M][N].
// ---------------------------------------------------------------------------
__global__ __launch_bounds__(256)
void gemm_bf16_xwT(const bf16_t* __restrict__ A, const float* __restrict__ W,
                   void* __restrict__ Cout, int M, int N, int K,
                   int f32out, int storeT)
{
  __shared__ bf16_t sA[64][32];
  __shared__ bf16_t sB[128][32];

  const int tid   = threadIdx.x;
  const int lane  = tid & 31;
  const int wave  = tid >> 5;
  const int hi    = lane >> 4;   // lane half (ISA A/B fragment layout)
  const int nl    = lane & 15;
  const int waveM = wave & 1;    // 0..1  -> 2 x 32 rows
  const int waveN = wave >> 1;   // 0..3  -> 4 x 32 cols
  const int m0    = blockIdx.y * 64;
  const int n0    = blockIdx.x * 128;

  // W staging coordinates
  const int brow = tid >> 1;        // 0..127
  const int bcol = (tid & 1) * 16;  // 0,16 (16 floats per thread)
#if !USE_TDM
  const int arow = tid >> 2;        // 0..63
  const int acol = (tid & 3) * 8;   // 0,8,16,24 (16B per thread)
#endif

#if USE_TDM
  // TDM descriptor group 1 (invariant): data_size=2B, tile 32x8,
  // tensor dims KxM, dim0 stride = K elements.  ISA 8.4 bit layout.
  v8i g1;
  g1[0] = 0x00010000;                                     // data_size = 1 -> 2B
  g1[1] = (int)(((unsigned)K & 0xffffu) << 16);           // tensor_dim0 [63:48]
  g1[2] = (int)(((unsigned)K >> 16) | (((unsigned)M & 0xffffu) << 16)); // [95:64]
  g1[3] = (int)(((unsigned)M >> 16) | (32u << 16));       // tile_dim0=32 [127:112]
  g1[4] = 8;                                              // tile_dim1=8 [143:128]
  g1[5] = (int)K;                                         // dim0_stride lo [191:160]
  g1[6] = 0;
  g1[7] = 0;
  const v4i gz4 = {};
  const v8i gz8 = {};
  const unsigned ldsA = (unsigned)(uintptr_t)&sA[0][0] + (unsigned)(wave * 8 * 32 * 2);
  const bf16_t* aSrc = A + (size_t)(m0 + wave * 8) * K;
#endif

  v8f c[2][2] = {};

  for (int k0 = 0; k0 < K; k0 += 32) {
    __syncthreads();   // previous tile fully consumed

#if USE_TDM
    {  // each wave DMAs its 8 rows of the A tile straight into LDS
      const unsigned long long ga = (unsigned long long)(uintptr_t)(aSrc + k0);
      v4u g0;
      g0[0] = 1u;                                   // count=1, user mode
      g0[1] = ldsA;                                 // lds_addr [63:32]
      g0[2] = (unsigned)ga;                         // global_addr lo
      g0[3] = (unsigned)(ga >> 32) | 0x80000000u;   // ga hi | type=2 [127:126]
      __builtin_amdgcn_tensor_load_to_lds(g0, g1, gz4, gz4, gz8, 0);
    }
#else
    *(uint4*)&sA[arow][acol] =
        *(const uint4*)&A[(size_t)(m0 + arow) * K + k0 + acol];
#endif

    // W tile: fp32 -> bf16 on the way into LDS (overlaps the TDM transfer)
    const float4* wp = (const float4*)&W[(size_t)(n0 + brow) * K + k0 + bcol];
    #pragma unroll
    for (int v = 0; v < 4; ++v) {
      float4 f = wp[v];
      sB[brow][bcol + v * 4 + 0] = (bf16_t)f.x;
      sB[brow][bcol + v * 4 + 1] = (bf16_t)f.y;
      sB[brow][bcol + v * 4 + 2] = (bf16_t)f.z;
      sB[brow][bcol + v * 4 + 3] = (bf16_t)f.w;
    }
    if (k0 + 32 < K) {  // global_prefetch_b8 of next W K-slab
      __builtin_prefetch(&W[(size_t)(n0 + brow) * K + k0 + 32 + bcol], 0, 1);
    }

#if USE_TDM
    __builtin_amdgcn_s_wait_tensorcnt(0);  // this wave's DMA slice landed
#endif
    __syncthreads();

    // A fragments: 16x32 bf16; lane = M (both halves), K split per lane half
    v16bf afrag[2], bfrag[2];
    #pragma unroll
    for (int i = 0; i < 2; ++i) {
      const int m  = waveM * 32 + i * 16 + nl;
      const int kh = hi * 16;
      #pragma unroll
      for (int e = 0; e < 8; ++e) {
        afrag[i][e]     = sA[m][kh + e];
        afrag[i][8 + e] = sA[m][kh + 8 + e];
      }
    }
    // B fragments: 32x16; lane = N, 16 contiguous K per lane half
    #pragma unroll
    for (int j = 0; j < 2; ++j) {
      const int n = waveN * 32 + j * 16 + nl;
      bfrag[j] = *(const v16bf*)&sB[n][hi * 16];
    }
    #pragma unroll
    for (int i = 0; i < 2; ++i)
      #pragma unroll
      for (int j = 0; j < 2; ++j)
        c[i][j] = __builtin_amdgcn_wmma_f32_16x16x32_bf16(
            false, afrag[i], false, bfrag[j], (short)0, c[i][j], false, false);
  }

  // C layout: lane = N, elem r = row (+8 for upper lane half)
  #pragma unroll
  for (int i = 0; i < 2; ++i)
    #pragma unroll
    for (int j = 0; j < 2; ++j)
      #pragma unroll
      for (int r = 0; r < 8; ++r) {
        const int m = m0 + waveM * 32 + i * 16 + hi * 8 + r;
        const int n = n0 + waveN * 32 + j * 16 + nl;
        const float val = c[i][j][r];
        if (storeT)       ((bf16_t*)Cout)[(size_t)n * M + m] = (bf16_t)val;
        else if (f32out)  ((float*)Cout)[(size_t)m * N + n]  = val;
        else              ((bf16_t*)Cout)[(size_t)m * N + n] = (bf16_t)val;
      }
}

// ---------------------------------------------------------------------------
// In-place RoPE on Q (T x H x D) and K (T x HKV x D), half = 64
// ---------------------------------------------------------------------------
__global__ void rope_kernel(bf16_t* __restrict__ Q, bf16_t* __restrict__ Kb) {
  const int qTotal = T_SEQ * HEADS * (DHEAD / 2);
  const int kTotal = T_SEQ * HKV_N * (DHEAD / 2);
  int idx = blockIdx.x * blockDim.x + threadIdx.x;
  if (idx >= qTotal + kTotal) return;

  bf16_t* buf; int ld, t, h, i;
  int x = idx;
  if (x < qTotal) { buf = Q;  ld = INNER_N;  i = x & 63; x >>= 6; h = x & 31; t = x >> 5; }
  else { x -= qTotal; buf = Kb; ld = KV_INNER; i = x & 63; x >>= 6; h = x & 7;  t = x >> 3; }

  // inv_freq = 10000^(-2i/D); ln(10000) = 9.2103403...
  const float inv = __expf(-((float)(2 * i) / (float)DHEAD) * 9.210340371976184f);
  const float ang = (float)t * inv;
  float sn, cs;
  __sincosf(ang, &sn, &cs);

  bf16_t* p = buf + (size_t)t * ld + h * DHEAD + i;
  const float x1 = (float)p[0];
  const float x2 = (float)p[64];
  p[0]  = (bf16_t)(x1 * cs - x2 * sn);
  p[64] = (bf16_t)(x2 * cs + x1 * sn);
}

// ---------------------------------------------------------------------------
// Causal flash attention. One wave per (head, 16-query tile).
// Q: [T][INNER] bf16 (RoPEd), Kb: [T][KV_INNER] bf16 (RoPEd),
// Vt: [KV_INNER][T] bf16 (transposed), O: [T][INNER] bf16.
// ---------------------------------------------------------------------------
__global__ __launch_bounds__(128)
void flash_attn_kernel(const bf16_t* __restrict__ Q, const bf16_t* __restrict__ Kb,
                       const bf16_t* __restrict__ Vt, bf16_t* __restrict__ O)
{
  __shared__ bf16_t sP[4][16][32];  // per-wave P repack scratch

  const int lane = threadIdx.x & 31;
  const int wave = threadIdx.x >> 5;
  const int hi   = lane >> 4;
  const int nl   = lane & 15;
  const int tile = blockIdx.x * 4 + wave;
  const int h    = tile >> 7;            // / (T/16)
  const int q0   = (tile & 127) * 16;
  const int hkv  = h >> 2;               // / Q_GROUP
  const float scale = 0.08838834764831845f;  // 1/sqrt(128)

  // Resident Q fragments: 16x128 bf16 as 4 A-fragments (16x32 each)
  v16bf qf[4];
  {
    const bf16_t* qrow = Q + (size_t)(q0 + nl) * INNER_N + h * DHEAD;
    const int kh = hi * 16;
    #pragma unroll
    for (int d = 0; d < 4; ++d)
      #pragma unroll
      for (int e = 0; e < 8; ++e) {
        qf[d][e]     = qrow[d * 32 + kh + e];
        qf[d][8 + e] = qrow[d * 32 + kh + 8 + e];
      }
  }

  v8f acc[8] = {};
  float mrow[8], lrow[8];
  #pragma unroll
  for (int r = 0; r < 8; ++r) { mrow[r] = -3.0e38f; lrow[r] = 0.f; }

  const int nChunks = (q0 + 16 + 31) >> 5;  // keys in chunks of 32, causal
  for (int kc = 0; kc < nChunks; ++kc) {
    const int keyBase = kc * 32;

    // S = Q * K^T : two 16x16 tiles, contraction over D=128 in 4 steps
    v8f s0 = {}, s1 = {};
    #pragma unroll
    for (int d = 0; d < 4; ++d) {
      const bf16_t* kp0 = Kb + (size_t)(keyBase + nl)      * KV_INNER + hkv * DHEAD + d * 32 + hi * 16;
      const bf16_t* kp1 = Kb + (size_t)(keyBase + 16 + nl) * KV_INNER + hkv * DHEAD + d * 32 + hi * 16;
      v16bf b0 = *(const v16bf*)kp0;
      v16bf b1 = *(const v16bf*)kp1;
      s0 = __builtin_amdgcn_wmma_f32_16x16x32_bf16(false, qf[d], false, b0, (short)0, s0, false, false);
      s1 = __builtin_amdgcn_wmma_f32_16x16x32_bf16(false, qf[d], false, b1, (short)0, s1, false, false);
    }

    // Online softmax. Row r lives in elem r across a 16-lane half.
    float alpha[8];
    #pragma unroll
    for (int r = 0; r < 8; ++r) {
      const int qrow = q0 + hi * 8 + r;
      float v0 = (keyBase + nl      <= qrow) ? s0[r] * scale : NEGBIG;
      float v1 = (keyBase + 16 + nl <= qrow) ? s1[r] * scale : NEGBIG;
      float mx = fmaxf(v0, v1);
      #pragma unroll
      for (int off = 8; off >= 1; off >>= 1) mx = fmaxf(mx, __shfl_xor(mx, off, 32));
      const float mnew = fmaxf(mrow[r], mx);
      alpha[r] = __expf(mrow[r] - mnew);
      const float p0 = __expf(v0 - mnew);
      const float p1 = __expf(v1 - mnew);
      float ps = p0 + p1;
      #pragma unroll
      for (int off = 8; off >= 1; off >>= 1) ps += __shfl_xor(ps, off, 32);
      lrow[r] = lrow[r] * alpha[r] + ps;
      mrow[r] = mnew;
      sP[wave][hi * 8 + r][nl]      = (bf16_t)p0;
      sP[wave][hi * 8 + r][16 + nl] = (bf16_t)p1;
      #pragma unroll
      for (int t = 0; t < 8; ++t) acc[t][r] *= alpha[r];
    }
    // wave-local LDS RAW: wait only on the DS split counter (CDNA5)
    asm volatile("s_wait_dscnt 0" ::: "memory");

    // Repack P as an A-fragment (16x32)
    v16bf pf;
    {
      const int kh = hi * 16;
      #pragma unroll
      for (int e = 0; e < 8; ++e) {
        pf[e]     = sP[wave][nl][kh + e];
        pf[8 + e] = sP[wave][nl][kh + 8 + e];
      }
    }
    // O += P * V : Vt rows are contiguous in key index -> clean B-fragments
    #pragma unroll
    for (int t = 0; t < 8; ++t) {
      const bf16_t* vp = Vt + (size_t)(hkv * DHEAD + t * 16 + nl) * T_SEQ + keyBase + hi * 16;
      v16bf vb = *(const v16bf*)vp;
      acc[t] = __builtin_amdgcn_wmma_f32_16x16x32_bf16(false, pf, false, vb, (short)0, acc[t], false, false);
    }
  }

  // Normalize and store attention output
  #pragma unroll
  for (int t = 0; t < 8; ++t)
    #pragma unroll
    for (int r = 0; r < 8; ++r) {
      const float o = acc[t][r] / lrow[r];
      O[(size_t)(q0 + hi * 8 + r) * INNER_N + h * DHEAD + t * 16 + nl] = (bf16_t)o;
    }
}

// ---------------------------------------------------------------------------
extern "C" void kernel_launch(void* const* d_in, const int* in_sizes, int n_in,
                              void* d_out, int out_size, void* d_ws, size_t ws_size,
                              hipStream_t stream) {
  const float* stm = (const float*)d_in[0];
  const float* w_q = (const float*)d_in[1];
  const float* w_k = (const float*)d_in[2];
  const float* w_v = (const float*)d_in[3];
  const float* w_o = (const float*)d_in[4];
  float* out = (float*)d_out;

  char* ws = (char*)d_ws;
  bf16_t* Xb = (bf16_t*)ws; ws += (size_t)T_SEQ * INNER_N * 2;   // x in bf16
  bf16_t* Qb = (bf16_t*)ws; ws += (size_t)T_SEQ * INNER_N * 2;   // Q
  bf16_t* Kb = (bf16_t*)ws; ws += (size_t)T_SEQ * KV_INNER * 2;  // K
  bf16_t* Vt = (bf16_t*)ws; ws += (size_t)KV_INNER * T_SEQ * 2;  // V transposed
  bf16_t* Oa = (bf16_t*)ws; ws += (size_t)T_SEQ * INNER_N * 2;   // attn out

  { int n = T_SEQ * INNER_N;
    cast_f32_bf16<<<(n + 255) / 256, 256, 0, stream>>>(stm, Xb, n); }

  // Q/K/V projections (V stored transposed for the P*V B-fragments)
  gemm_bf16_xwT<<<dim3(INNER_N / 128,  T_SEQ / 64), 256, 0, stream>>>(Xb, w_q, Qb, T_SEQ, INNER_N,  INNER_N, 0, 0);
  gemm_bf16_xwT<<<dim3(KV_INNER / 128, T_SEQ / 64), 256, 0, stream>>>(Xb, w_k, Kb, T_SEQ, KV_INNER, INNER_N, 0, 0);
  gemm_bf16_xwT<<<dim3(KV_INNER / 128, T_SEQ / 64), 256, 0, stream>>>(Xb, w_v, Vt, T_SEQ, KV_INNER, INNER_N, 0, 1);

  { int n = T_SEQ * HEADS * (DHEAD / 2) + T_SEQ * HKV_N * (DHEAD / 2);
    rope_kernel<<<(n + 255) / 256, 256, 0, stream>>>(Qb, Kb); }

  flash_attn_kernel<<<dim3(HEADS * (T_SEQ / 16) / 4), 128, 0, stream>>>(Qb, Kb, Vt, Oa);

  // Output projection, fp32 result straight into d_out
  gemm_bf16_xwT<<<dim3(INNER_N / 128, T_SEQ / 64), 256, 0, stream>>>(Oa, w_o, out, T_SEQ, INNER_N, INNER_N, 1, 0);
}
